// OffsetBlock_68049461838663
// MI455X (gfx1250) — matrix operand order repaired and checked by reading, not compiled, not run on hardware
//
#include <hip/hip_runtime.h>

// ---------------- types ----------------
typedef __attribute__((ext_vector_type(16))) __bf16 v16bf;
typedef __attribute__((ext_vector_type(8)))  float  v8f;

union V16U { v16bf v; uint4 q[2]; };
union U4x8 { uint4 q; unsigned short s[8]; };

__device__ inline unsigned short f2bf(float f) {
    unsigned u = __float_as_uint(f);
    u += 0x7FFFu + ((u >> 16) & 1u);          // round-to-nearest-even
    return (unsigned short)(u >> 16);
}
__device__ inline float bf2f(unsigned short h) {
    return __uint_as_float(((unsigned)h) << 16);
}

// Block reduction (256 threads)
__device__ inline float block_sum256(float v, float* buf) {
    int t = threadIdx.x;
    buf[t] = v; __syncthreads();
    for (int o = 128; o > 0; o >>= 1) {
        if (t < o) buf[t] += buf[t + o];
        __syncthreads();
    }
    return buf[0];
}

#define HW   16384
#define LG2  14

// =====================================================================
// K1: fused conv1x1 for pre (rows 0..127) and atten (rows 128..255)
//   C[256 x N] = W[256x64] * X[64x N], bf16 WMMA, f32 acc
// Weights persist in LDS; block loops over 8 chunks of 32 pixels.
// block = 256 thr (8 waves); wave w -> M rows [32w,32w+32), N = 32
// =====================================================================
__global__ __launch_bounds__(256) void k_conv_pre_atten(
    const float* __restrict__ x,
    const float* __restrict__ pre_w, const float* __restrict__ pre_b,
    const float* __restrict__ atten_w, const float* __restrict__ atten_b,
    unsigned short* __restrict__ outpre, unsigned short* __restrict__ araw)
{
    __shared__ unsigned short Wl[256 * 64];   // [M][K] row-major, 32 KB
    __shared__ unsigned short Xl[32 * 64];    // [pixel][K] (transposed), 4 KB
    __shared__ float          Bl[256];

    const int t = threadIdx.x;
    const int b = blockIdx.y;

    // stacked weights + bias -> LDS (bf16), vectorized
    {
        const float4* src = (const float4*)((t < 128) ? (pre_w + t * 64)
                                                      : (atten_w + (t - 128) * 64));
#pragma unroll
        for (int kk = 0; kk < 16; ++kk) {
            float4 f = src[kk];
            Wl[t * 64 + kk * 4 + 0] = f2bf(f.x);
            Wl[t * 64 + kk * 4 + 1] = f2bf(f.y);
            Wl[t * 64 + kk * 4 + 2] = f2bf(f.z);
            Wl[t * 64 + kk * 4 + 3] = f2bf(f.w);
        }
        Bl[t] = (t < 128) ? pre_b[t] : atten_b[t - 128];
    }

    const int wave = t >> 5, lane = t & 31;
    const int lan15 = lane & 15, hi = lane >> 4;
    const int ci = t >> 2;
    const int p0 = (t & 3) * 8;

    for (int chunk = 0; chunk < 8; ++chunk) {
        const int pb = blockIdx.x * 256 + chunk * 32;

        __syncthreads();  // Xl consumers from previous chunk done (also covers Wl 1st time)
        {   // x tile -> LDS transposed [pixel][ci]
            const float4* xs = (const float4*)(x + (((size_t)b * 64 + ci) << LG2) + pb + p0);
            float4 f0 = xs[0], f1 = xs[1];
            Xl[(p0 + 0) * 64 + ci] = f2bf(f0.x);
            Xl[(p0 + 1) * 64 + ci] = f2bf(f0.y);
            Xl[(p0 + 2) * 64 + ci] = f2bf(f0.z);
            Xl[(p0 + 3) * 64 + ci] = f2bf(f0.w);
            Xl[(p0 + 4) * 64 + ci] = f2bf(f1.x);
            Xl[(p0 + 5) * 64 + ci] = f2bf(f1.y);
            Xl[(p0 + 6) * 64 + ci] = f2bf(f1.z);
            Xl[(p0 + 7) * 64 + ci] = f2bf(f1.w);
        }
        __syncthreads();

        v8f acc[2][2];
#pragma unroll
        for (int mi = 0; mi < 2; ++mi)
#pragma unroll
            for (int ni = 0; ni < 2; ++ni)
#pragma unroll
                for (int r = 0; r < 8; ++r) acc[mi][ni][r] = 0.0f;

#pragma unroll
        for (int ks = 0; ks < 64; ks += 32) {
            V16U A[2], Bv[2];
            const int klo = hi * 8;    // A: split-K halves across half-waves
            const int kh  = hi * 8;    // B: split-K halves across half-waves (same as A)
#pragma unroll
            for (int mi = 0; mi < 2; ++mi) {
                int row = wave * 32 + mi * 16 + lan15;
                const uint4* p = (const uint4*)&Wl[row * 64 + ks + klo];
                A[mi].q[0] = p[0];
                A[mi].q[1] = p[2];     // +16 ushorts (K += 16)
            }
#pragma unroll
            for (int ni = 0; ni < 2; ++ni) {
                int pix = ni * 16 + lan15;
                const uint4* p = (const uint4*)&Xl[pix * 64 + ks + kh];
                Bv[ni].q[0] = p[0];
                Bv[ni].q[1] = p[2];
            }
#pragma unroll
            for (int mi = 0; mi < 2; ++mi)
#pragma unroll
                for (int ni = 0; ni < 2; ++ni)
                    acc[mi][ni] = __builtin_amdgcn_wmma_f32_16x16x32_bf16(
                        false, A[mi].v, false, Bv[ni].v, (short)0, acc[mi][ni], false, false);
        }

        // epilogue: C layout VGPR r -> M = r + 8*(lane>=16), N = lane&15
#pragma unroll
        for (int mi = 0; mi < 2; ++mi)
#pragma unroll
            for (int ni = 0; ni < 2; ++ni)
#pragma unroll
                for (int r = 0; r < 8; ++r) {
                    int mg = wave * 32 + mi * 16 + r + hi * 8;
                    int pg = pb + ni * 16 + lan15;
                    float v = acc[mi][ni][r] + Bl[mg];
                    size_t idx = (((size_t)b * 128 + (mg & 127)) << LG2) + pg;
                    if (mg < 128) outpre[idx] = f2bf(v);
                    else          araw[idx]   = f2bf(v);
                }
    }
}

// =====================================================================
// K2: per-channel sum / sumsq (bf16 / f32 variants)
// =====================================================================
__global__ __launch_bounds__(256) void k_stats_bf16(
    const unsigned short* __restrict__ a, float* __restrict__ S, float* __restrict__ S2)
{
    __shared__ float r1[256], r2[256];
    int c = blockIdx.x, b = blockIdx.y, C = gridDim.x;
    const unsigned short* p = a + (((size_t)b * C + c) << LG2);
    float s = 0.f, s2 = 0.f;
    for (int i = threadIdx.x; i < HW; i += 256) {
        __builtin_prefetch(p + i + 4096, 0, 1);
        float v = bf2f(p[i]); s += v; s2 += v * v;
    }
    float ts  = block_sum256(s,  r1);
    float ts2 = block_sum256(s2, r2);
    if (threadIdx.x == 0) { atomicAdd(&S[c], ts); atomicAdd(&S2[c], ts2); }
}

__global__ __launch_bounds__(256) void k_stats_f32(
    const float* __restrict__ a, float* __restrict__ S, float* __restrict__ S2)
{
    __shared__ float r1[256], r2[256];
    int c = blockIdx.x, b = blockIdx.y, C = gridDim.x;
    const float* p = a + (((size_t)b * C + c) << LG2);
    float s = 0.f, s2 = 0.f;
    for (int i = threadIdx.x; i < HW; i += 256) {
        __builtin_prefetch(p + i + 4096, 0, 1);
        float v = p[i]; s += v; s2 += v * v;
    }
    float ts  = block_sum256(s,  r1);
    float ts2 = block_sum256(s2, r2);
    if (threadIdx.x == 0) { atomicAdd(&S[c], ts); atomicAdd(&S2[c], ts2); }
}

// mean/var -> affine scale/shift
__global__ void k_bn_finalize(const float* __restrict__ S, const float* __restrict__ S2,
                              const float* __restrict__ g, const float* __restrict__ bta,
                              float* __restrict__ scale, float* __restrict__ shift,
                              int C, float invN)
{
    int c = threadIdx.x;
    if (c < C) {
        float mean = S[c] * invN;
        float var  = S2[c] * invN - mean * mean;
        float sc   = g[c] * rsqrtf(var + 1e-5f);
        scale[c] = sc;
        shift[c] = bta[c] - mean * sc;
    }
}

// =====================================================================
// K3a: s = softplus(bn(a)) in place (bf16) + per-(b,c) spatial sum
// =====================================================================
__global__ __launch_bounds__(256) void k_softplus_sum(
    unsigned short* __restrict__ a, const float* __restrict__ scale,
    const float* __restrict__ shift, float* __restrict__ sumS)
{
    __shared__ float r1[256];
    int c = blockIdx.x, b = blockIdx.y;
    unsigned short* p = a + (((size_t)b * 128 + c) << LG2);
    float sc = scale[c], sh = shift[c], s = 0.f;
    for (int i = threadIdx.x; i < HW; i += 256) {
        __builtin_prefetch(p + i + 4096, 0, 1);
        float v  = sc * bf2f(p[i]) + sh;
        float sp = (v > 20.f) ? v : log1pf(expf(v));
        p[i] = f2bf(sp);
        s += sp;
    }
    float ts = block_sum256(s, r1);
    if (threadIdx.x == 0) sumS[b * 128 + c] = ts;
}

// =====================================================================
// K3b: m = (s / sum_s) * bilinear_displace(out_pre)   (bf16 out)
// =====================================================================
__global__ __launch_bounds__(256) void k_displace_mul(
    const unsigned short* __restrict__ pre, const unsigned short* __restrict__ s,
    const float* __restrict__ offsets, const float* __restrict__ sumS,
    unsigned short* __restrict__ m)
{
    int c = blockIdx.x, b = blockIdx.y;
    float dy = offsets[(c >> 2) * 2 + 0];
    float dx = offsets[(c >> 2) * 2 + 1];
    float fy = floorf(dy), fx = floorf(dx);
    int   iy = (int)fy,    ix = (int)fx;
    float wy = dy - fy,    wx = dx - fx;
    float w00 = (1.f - wy) * (1.f - wx), w01 = (1.f - wy) * wx;
    float w10 = wy * (1.f - wx),         w11 = wy * wx;

    size_t base = ((size_t)b * 128 + c) << LG2;
    float inv = 1.f / sumS[b * 128 + c];

    for (int i = threadIdx.x; i < HW; i += 256) {
        int h = i >> 7, w = i & 127;
        auto tap = [&](int oy, int ox) -> float {
            int ys = h + oy, xs = w + ox;
            if (ys < 0 || ys > 127 || xs < 0 || xs > 127) return 0.f;
            return bf2f(pre[base + (ys << 7) + xs]);
        };
        float d = w00 * tap(iy, ix)     + w01 * tap(iy, ix + 1)
                + w10 * tap(iy + 1, ix) + w11 * tap(iy + 1, ix + 1);
        float val = bf2f(s[base + i]) * inv * d;
        m[base + i] = f2bf(val);
    }
}

// =====================================================================
// K4: out_skip = x + conv1x1(m, post_w) + post_b   (K=128, M=64)
// Weights persist in LDS; block loops over 8 chunks of 32 pixels.
// block: 256 thr = 8 waves; wave w -> (mi = w&3, ni = w>>2) 16x16 tile
// =====================================================================
__global__ __launch_bounds__(256) void k_conv_post_skip(
    const unsigned short* __restrict__ m,
    const float* __restrict__ post_w, const float* __restrict__ post_b,
    const float* __restrict__ x, float* __restrict__ skip)
{
    __shared__ unsigned short Wl[64 * 128];   // [M][K], 16 KB
    __shared__ unsigned short Ml[32 * 128];   // [pixel][K] transposed, 8 KB
    __shared__ float          Bl[64];

    const int t = threadIdx.x;
    const int b = blockIdx.y;

    {
        int row = t >> 2, k0 = (t & 3) * 32;
        const float4* src = (const float4*)(post_w + row * 128 + k0);
#pragma unroll
        for (int kk = 0; kk < 8; ++kk) {
            float4 f = src[kk];
            Wl[row * 128 + k0 + kk * 4 + 0] = f2bf(f.x);
            Wl[row * 128 + k0 + kk * 4 + 1] = f2bf(f.y);
            Wl[row * 128 + k0 + kk * 4 + 2] = f2bf(f.z);
            Wl[row * 128 + k0 + kk * 4 + 3] = f2bf(f.w);
        }
        if (t < 64) Bl[t] = post_b[t];
    }

    const int wave = t >> 5, lane = t & 31;
    const int lan15 = lane & 15, hi = lane >> 4;
    const int mi = wave & 3, ni = wave >> 2;
    const int cd = t >> 1;
    const int p0 = (t & 1) * 16;

    for (int chunk = 0; chunk < 8; ++chunk) {
        const int pb = blockIdx.x * 256 + chunk * 32;

        __syncthreads();
        {   // m tile -> LDS transposed [pixel][cd], vectorized global reads
            const uint4* src = (const uint4*)(m + (((size_t)b * 128 + cd) << LG2) + pb + p0);
            U4x8 u0, u1;
            u0.q = src[0];
            u1.q = src[1];
#pragma unroll
            for (int i = 0; i < 8; ++i) Ml[(p0 + i)     * 128 + cd] = u0.s[i];
#pragma unroll
            for (int i = 0; i < 8; ++i) Ml[(p0 + 8 + i) * 128 + cd] = u1.s[i];
        }
        __syncthreads();

        v8f acc;
#pragma unroll
        for (int r = 0; r < 8; ++r) acc[r] = 0.0f;

#pragma unroll
        for (int ks = 0; ks < 128; ks += 32) {
            V16U A, Bv;
            int row = mi * 16 + lan15;
            const uint4* pa = (const uint4*)&Wl[row * 128 + ks + hi * 8];
            A.q[0] = pa[0];
            A.q[1] = pa[2];
            int pix = ni * 16 + lan15;
            const uint4* pv = (const uint4*)&Ml[pix * 128 + ks + hi * 8];
            Bv.q[0] = pv[0];
            Bv.q[1] = pv[2];
            acc = __builtin_amdgcn_wmma_f32_16x16x32_bf16(
                false, A.v, false, Bv.v, (short)0, acc, false, false);
        }

#pragma unroll
        for (int r = 0; r < 8; ++r) {
            int mg = mi * 16 + r + hi * 8;
            int pg = pb + ni * 16 + lan15;
            size_t idx = (((size_t)b * 64 + mg) << LG2) + pg;
            skip[idx] = acc[r] + Bl[mg] + x[idx];
        }
    }
}

// =====================================================================
// K6: out = relu(bn(skip))
// =====================================================================
__global__ __launch_bounds__(256) void k_bn_relu(
    const float* __restrict__ skip, const float* __restrict__ scale,
    const float* __restrict__ shift, float* __restrict__ out)
{
    size_t i = (size_t)blockIdx.x * 256 + threadIdx.x;
    int c = (int)((i >> LG2) & 63);
    float v = scale[c] * skip[i] + shift[c];
    out[i] = v > 0.f ? v : 0.f;
}

// =====================================================================
extern "C" void kernel_launch(void* const* d_in, const int* in_sizes, int n_in,
                              void* d_out, int out_size, void* d_ws, size_t ws_size,
                              hipStream_t stream)
{
    (void)in_sizes; (void)n_in; (void)out_size; (void)ws_size;

    const float* x          = (const float*)d_in[0];
    const float* pre_w      = (const float*)d_in[1];
    const float* pre_b      = (const float*)d_in[2];
    const float* post_w     = (const float*)d_in[3];
    const float* post_b     = (const float*)d_in[4];
    const float* atten_w    = (const float*)d_in[5];
    const float* atten_b    = (const float*)d_in[6];
    const float* atten_bn_g = (const float*)d_in[7];
    const float* atten_bn_b = (const float*)d_in[8];
    const float* offsets    = (const float*)d_in[9];
    const float* bn_g       = (const float*)d_in[10];
    const float* bn_b       = (const float*)d_in[11];

    char* ws = (char*)d_ws;
    unsigned short* pre  = (unsigned short*)(ws);                          // 64 MB bf16
    unsigned short* a    = (unsigned short*)(ws + ((size_t)64  << 20));    // 64 MB bf16 (a, then s)
    unsigned short* mm   = (unsigned short*)(ws + ((size_t)128 << 20));    // 64 MB bf16
    float*          skip = (float*)         (ws + ((size_t)192 << 20));    // 64 MB f32
    float*          st   = (float*)         (ws + ((size_t)256 << 20));    // stats

    float* attenS     = st;
    float* attenS2    = st + 128;
    float* attenScale = st + 256;
    float* attenShift = st + 384;
    float* sumS       = st + 512;        // 16*128
    float* skipS      = st + 2560;
    float* skipS2     = st + 2624;
    float* skipScale  = st + 2688;
    float* skipShift  = st + 2752;

    hipMemsetAsync(st, 0, 2816 * sizeof(float), stream);

    const float invN = 1.0f / (16.0f * 16384.0f);

    k_conv_pre_atten<<<dim3(64, 16), 256, 0, stream>>>(
        x, pre_w, pre_b, atten_w, atten_b, pre, a);

    k_stats_bf16<<<dim3(128, 16), 256, 0, stream>>>(a, attenS, attenS2);
    k_bn_finalize<<<1, 128, 0, stream>>>(attenS, attenS2, atten_bn_g, atten_bn_b,
                                         attenScale, attenShift, 128, invN);

    k_softplus_sum<<<dim3(128, 16), 256, 0, stream>>>(a, attenScale, attenShift, sumS);

    k_displace_mul<<<dim3(128, 16), 256, 0, stream>>>(pre, a, offsets, sumS, mm);

    k_conv_post_skip<<<dim3(64, 16), 256, 0, stream>>>(mm, post_w, post_b, x, skip);

    k_stats_f32<<<dim3(64, 16), 256, 0, stream>>>(skip, skipS, skipS2);
    k_bn_finalize<<<1, 64, 0, stream>>>(skipS, skipS2, bn_g, bn_b,
                                        skipScale, skipShift, 64, invN);

    k_bn_relu<<<65536, 256, 0, stream>>>(skip, skipScale, skipShift, (float*)d_out);
}
